// DayEncoder_21157008900560
// MI455X (gfx1250) — matrix operand order repaired and checked by reading, not compiled, not run on hardware
//
#include <hip/hip_runtime.h>
#include <hip/hip_bf16.h>

// ---- model constants -------------------------------------------------------
#define DM     512      // d_model
#define DI     512      // d_inner
#define DSTATE 64
#define RNK    32       // dt_rank
#define NL     12
#define NB     16       // batch
#define NEP    1020
#define NSUM   4
#define LSEQ   1024     // NSUM + NEP
#define BL     (NB * LSEQ)   // 16384 rows
#define MEP    (NB * NEP)    // 16320 rows
#define XPAD   192           // x_proj N (160) padded to multiple of 64

// gfx1250 async global->LDS path (ch.08), guarded so host pass / older
// toolchains fall back to plain LDS stores. Probe round 2 revealed the
// builtin signature: (as1 int4* gsrc, as3 int4* ldsdst, imm offset, imm cpol).
#if defined(__AMDGCN__) && __has_builtin(__builtin_amdgcn_global_load_async_to_lds_b128) && __has_builtin(__builtin_amdgcn_s_wait_asynccnt)
#define HAVE_ASYNC_LDS 1
typedef __attribute__((ext_vector_type(4))) int v4i_t;
typedef __attribute__((address_space(1))) v4i_t* gptr_v4i;
typedef __attribute__((address_space(3))) v4i_t* lptr_v4i;
#endif

typedef __attribute__((ext_vector_type(16))) _Float16 v16h;
typedef __attribute__((ext_vector_type(8)))  _Float16 v8h;
typedef __attribute__((ext_vector_type(8)))  float    v8f;

union V16U { v16h v; v8h h[2]; };

// ---------------------------------------------------------------------------
// Generic wave-tile WMMA GEMM:  C[M,N] = A[M,K] * W[N,K]^T   (f16 in, f32 acc)
// Each wave32 computes a 16x64 strip with v_wmma_f32_16x16x32_f16.
// EPI: 0=none, 1=bias+GELU(exact), 2=bias, 3=softplus(bias+x), 4=C += acc
// ---------------------------------------------------------------------------
template <int EPI>
__global__ __launch_bounds__(256) void gemm16(
    const _Float16* __restrict__ A, const _Float16* __restrict__ W,
    const float* __restrict__ bias,
    float* __restrict__ Cf, _Float16* __restrict__ Ch,
    int M, int N, int K, int lda, int ldw, int ldc)
{
  const int wave   = blockIdx.x * (blockDim.x >> 5) + (threadIdx.x >> 5);
  const int lane   = threadIdx.x & 31;
  const int tilesN = N >> 6;
  const int tilesM = M >> 4;
  const int tM = wave / tilesN;
  const int tN = wave - tM * tilesN;
  if (tM >= tilesM) return;
  const int m0 = tM << 4, n0 = tN << 6;
  const int mr = lane & 15, hh = lane >> 4;

  // A fragment (ISA 7.12.2, 16-bit A 16x32): lane(m=mr, half=hh) holds
  // K = [8h..8h+7] in v0..3 and K = [16+8h..16+8h+7] in v4..7.
  const _Float16* Ap = A + (size_t)(m0 + mr) * lda + 8 * hh;
  // B fragment (K x 16 col-per-lane): lane n=mr holds K = [8h..8h+7] and [16+8h..16+8h+7].
  const _Float16* Wp = W + (size_t)(n0 + mr) * ldw + 8 * hh;

  v8f acc[4] = {};
  for (int k = 0; k < K; k += 32) {
    // keep L0/WGP$ warm 4 k-steps (256B) ahead -> global_prefetch_b8
    __builtin_prefetch(Ap + k + 128, 0, 3);
    __builtin_prefetch(Wp + k + 128, 0, 3);
    V16U a;
    a.h[0] = *(const v8h*)(Ap + k);
    a.h[1] = *(const v8h*)(Ap + k + 16);
#pragma unroll
    for (int t = 0; t < 4; ++t) {
      V16U bu;
      bu.h[0] = *(const v8h*)(Wp + (size_t)(16 * t) * ldw + k);
      bu.h[1] = *(const v8h*)(Wp + (size_t)(16 * t) * ldw + k + 16);
      v16h b = bu.v;
      acc[t] = __builtin_amdgcn_wmma_f32_16x16x32_f16(
          false, a.v, false, b, (short)0, acc[t], false, false);
    }
  }

#pragma unroll
  for (int t = 0; t < 4; ++t) {
    const int col = n0 + 16 * t + mr;
    const float bv = bias ? bias[col] : 0.0f;
#pragma unroll
    for (int r = 0; r < 8; ++r) {
      const int row = m0 + 8 * hh + r;        // C/D: VGPR r -> M = 8*half + r
      float v = acc[t][r];
      if (EPI == 1 || EPI == 2) v += bv;
      if (EPI == 1) v = 0.5f * v * (1.0f + erff(v * 0.70710678118654752f));
      if (EPI == 3) { v += bv; v = (v > 20.0f) ? v : log1pf(__expf(v)); }
      const size_t idx = (size_t)row * ldc + col;
      if (EPI == 4)      { Cf[idx] += v; }
      else if (Cf)       { Cf[idx] = v; }
      if (Ch) Ch[idx] = (_Float16)v;
    }
  }
}

// ---- fp32 -> fp16 weight conversion ---------------------------------------
__global__ __launch_bounds__(256) void k_convert(
    const float* __restrict__ s, _Float16* __restrict__ d, long long n)
{
  long long i = (long long)blockIdx.x * blockDim.x + threadIdx.x;
  if (i < n) d[i] = (_Float16)s[i];
}

// x_proj_W (12,2,160,512) -> f16 padded to (12,2,192,512), pad rows zero.
__global__ __launch_bounds__(256) void k_pad_xproj(
    const float* __restrict__ s, _Float16* __restrict__ d)
{
  long long i = (long long)blockIdx.x * blockDim.x + threadIdx.x;
  const long long total = (long long)NL * 2 * XPAD * DM;
  if (i >= total) return;
  int col  = (int)(i % DM);
  long long rr = i / DM;
  int row  = (int)(rr % XPAD);
  long long ld = rr / XPAD;              // l*2+dir
  d[i] = (row < 160) ? (_Float16)s[(ld * 160 + row) * DM + col] : (_Float16)0.0f;
}

// concat(waveform, rhythm) -> f16 [MEP, 512]
__global__ __launch_bounds__(256) void k_concat(
    const float* __restrict__ wav, const float* __restrict__ rhy,
    _Float16* __restrict__ cat)
{
  long long i = (long long)blockIdx.x * blockDim.x + threadIdx.x;
  if (i >= (long long)MEP * DM) return;
  int col = (int)(i % DM);
  long long row = i / DM;
  float v = (col < 384) ? wav[row * 384 + col] : rhy[row * 128 + (col - 384)];
  cat[i] = (_Float16)v;
}

// x[b,0:4] = summary tokens; x[b,4+t] = fused z[b,t]
__global__ __launch_bounds__(256) void k_build_x(
    const float* __restrict__ zfus, const float* __restrict__ summ,
    float* __restrict__ x)
{
  long long i = (long long)blockIdx.x * blockDim.x + threadIdx.x;
  if (i >= (long long)BL * DM) return;
  int col = (int)(i & (DM - 1));
  long long row = i >> 9;
  int pos = (int)(row & (LSEQ - 1));
  long long b = row >> 10;
  x[i] = (pos < NSUM) ? summ[pos * DM + col]
                      : zfus[(b * NEP + (pos - NSUM)) * DM + col];
}

// one wave32 per row layernorm; optional f32 and f16 outputs
__global__ __launch_bounds__(256) void k_layernorm(
    const float* __restrict__ X, const float* __restrict__ g,
    const float* __restrict__ bta, float* __restrict__ outf,
    _Float16* __restrict__ outh, int rows)
{
  const int row  = blockIdx.x * 8 + (threadIdx.x >> 5);
  if (row >= rows) return;
  const int lane = threadIdx.x & 31;
  const float* xr = X + (size_t)row * DM;
  float vals[16];
  float s = 0.0f;
#pragma unroll
  for (int i = 0; i < 16; ++i) { vals[i] = xr[lane + 32 * i]; s += vals[i]; }
#pragma unroll
  for (int o = 16; o > 0; o >>= 1) s += __shfl_xor(s, o, 32);
  const float mean = s * (1.0f / DM);
  float vv = 0.0f;
#pragma unroll
  for (int i = 0; i < 16; ++i) { float dd = vals[i] - mean; vv += dd * dd; }
#pragma unroll
  for (int o = 16; o > 0; o >>= 1) vv += __shfl_xor(vv, o, 32);
  const float rstd = rsqrtf(vv * (1.0f / DM) + 1e-5f);
#pragma unroll
  for (int i = 0; i < 16; ++i) {
    const int c = lane + 32 * i;
    const float v = (vals[i] - mean) * rstd * g[c] + bta[c];
    if (outf) outf[(size_t)row * DM + c] = v;
    if (outh) outh[(size_t)row * DM + c] = (_Float16)v;
  }
}

// depthwise causal (dir=0) / anti-causal (dir=1) conv + SiLU.
// dir=1 implements conv on the time-reversed sequence in original coords:
//   u[t] = b + sum_j w[j] * xz[t + 3 - j]
__global__ __launch_bounds__(256) void k_conv_silu(
    const float* __restrict__ xz, const float* __restrict__ cw,
    const float* __restrict__ cb, float* __restrict__ xcf,
    _Float16* __restrict__ xch, int dir)
{
  long long idx = (long long)blockIdx.x * blockDim.x + threadIdx.x;
  if (idx >= (long long)BL * DI) return;
  const int d   = (int)(idx & (DI - 1));
  const long long row = idx >> 9;
  const int t   = (int)(row & (LSEQ - 1));
  const long long base_b = row - t;  // b * LSEQ
  float acc = cb[d];
#pragma unroll
  for (int j = 0; j < 4; ++j) {
    const int tt = dir ? (t + 3 - j) : (t - 3 + j);
    if (tt >= 0 && tt < LSEQ)
      acc += cw[d * 4 + j] * xz[(base_b + tt) * 1024 + d];
  }
  const float s = acc / (1.0f + __expf(-acc));
  xcf[idx] = s;
  xch[idx] = (_Float16)s;
}

// selective scan: one lane owns one channel d; h[64], A[64] in VGPRs.
// Per step B_t,C_t (128 contiguous floats at dblf[row*192+32..159]) are
// staged into LDS; on gfx1250 via async global->LDS b128 + s_wait_asynccnt,
// tracked with ASYNCcnt (ISA ch.08), else via plain LDS stores.
__global__ __launch_bounds__(128) void k_scan(
    const float* __restrict__ dblf, const float* __restrict__ dtf,
    const float* __restrict__ xcf,  const float* __restrict__ xz,
    const float* __restrict__ A_log, const float* __restrict__ Dsk,
    _Float16* __restrict__ y16, int dir)
{
  __shared__ float sBC[2][2 * DSTATE];   // [buf][0:64)=B_t, [64:128)=C_t
  const int b = blockIdx.x;
  const int d = blockIdx.y * 128 + threadIdx.x;
  float Aa[DSTATE], hs[DSTATE];
  const float* al = A_log + (size_t)d * DSTATE;
#pragma unroll
  for (int n = 0; n < DSTATE; ++n) { Aa[n] = -__expf(al[n]); hs[n] = 0.0f; }
  const float Dp = Dsk[d];
  for (int s = 0; s < LSEQ; ++s) {
    const int t = dir ? (LSEQ - 1 - s) : s;
    const size_t row = (size_t)b * LSEQ + t;
    const int buf = s & 1;
#if defined(HAVE_ASYNC_LDS)
    if (threadIdx.x < 32) {                       // 32 lanes x 16B = 512B tile
      const float* g = dblf + row * XPAD + RNK + 4 * threadIdx.x;
      __builtin_amdgcn_global_load_async_to_lds_b128(
          (gptr_v4i)g, (lptr_v4i)&sBC[buf][4 * threadIdx.x], 0, 0);
      __builtin_amdgcn_s_wait_asynccnt(0);
    }
#else
    sBC[buf][threadIdx.x] = dblf[row * XPAD + RNK + threadIdx.x];
#endif
    __syncthreads();
    const float dt  = dtf[row * DI + d];
    const float xv  = xcf[row * DI + d];
    const float dtx = dt * xv;
    float acc = 0.0f;
#pragma unroll
    for (int n = 0; n < DSTATE; ++n) {
      const float dA = __expf(dt * Aa[n]);     // v_exp_f32 co-issues with FMA
      hs[n] = dA * hs[n] + dtx * sBC[buf][n];
      acc  += hs[n] * sBC[buf][DSTATE + n];
    }
    float y = acc + xv * Dp;
    const float zv = xz[row * 1024 + DI + d];
    y *= zv / (1.0f + __expf(-zv));            // y * silu(z)
    y16[row * DI + d] = (_Float16)y;
  }
}

// attention scores: one wave per row, dot(xln_row, attn_W) + b
__global__ __launch_bounds__(256) void k_scores(
    const float* __restrict__ xln, const float* __restrict__ aW,
    const float* __restrict__ ab, float* __restrict__ sc)
{
  const int row = blockIdx.x * 8 + (threadIdx.x >> 5);
  if (row >= BL) return;
  const int lane = threadIdx.x & 31;
  const float* xr = xln + (size_t)row * DM;
  float s = 0.0f;
  for (int i = lane; i < DM; i += 32) s += xr[i] * aW[i];
#pragma unroll
  for (int o = 16; o > 0; o >>= 1) s += __shfl_xor(s, o, 32);
  if (lane == 0) sc[row] = s + ab[0];
}

// per-batch softmax over L and weighted pooling -> day_embed[b, 512]
__global__ __launch_bounds__(256) void k_pool(
    const float* __restrict__ xln, const float* __restrict__ scores,
    float* __restrict__ day)
{
  __shared__ float p[LSEQ];
  __shared__ float red[256];
  const int b = blockIdx.x, tid = threadIdx.x;
  const float* sc = scores + (size_t)b * LSEQ;
  float m = -3.4e38f;
  for (int i = tid; i < LSEQ; i += 256) m = fmaxf(m, sc[i]);
  red[tid] = m; __syncthreads();
  for (int o = 128; o > 0; o >>= 1) { if (tid < o) red[tid] = fmaxf(red[tid], red[tid + o]); __syncthreads(); }
  m = red[0]; __syncthreads();
  float s = 0.0f;
  for (int i = tid; i < LSEQ; i += 256) { float e = __expf(sc[i] - m); p[i] = e; s += e; }
  red[tid] = s; __syncthreads();
  for (int o = 128; o > 0; o >>= 1) { if (tid < o) red[tid] += red[tid + o]; __syncthreads(); }
  const float inv = 1.0f / red[0];
  __syncthreads();
  for (int d = tid; d < DM; d += 256) {
    float acc = 0.0f;
    for (int i = 0; i < LSEQ; ++i) acc += p[i] * xln[((size_t)b * LSEQ + i) * DM + d];
    day[(size_t)b * DM + d] = acc * inv;
  }
}

// episode_ctx = xln[:, NSUM:, :]
__global__ __launch_bounds__(256) void k_ctx(
    const float* __restrict__ xln, float* __restrict__ out)
{
  long long i = (long long)blockIdx.x * blockDim.x + threadIdx.x;
  if (i >= (long long)MEP * DM) return;
  const int col = (int)(i % DM);
  const long long row = i / DM;
  const int t = (int)(row % NEP);
  const long long b = row / NEP;
  out[i] = xln[((size_t)b * LSEQ + NSUM + t) * DM + col];
}

// ---------------------------------------------------------------------------
extern "C" void kernel_launch(void* const* d_in, const int* in_sizes, int n_in,
                              void* d_out, int out_size, void* d_ws, size_t ws_size,
                              hipStream_t stream)
{
  (void)in_sizes; (void)n_in; (void)out_size; (void)ws_size;

  const float* wav    = (const float*)d_in[0];
  const float* rhy    = (const float*)d_in[1];
  const float* fW1    = (const float*)d_in[2];
  const float* fb1    = (const float*)d_in[3];
  const float* fW2    = (const float*)d_in[4];
  const float* fb2    = (const float*)d_in[5];
  const float* summ   = (const float*)d_in[6];
  const float* ln_g   = (const float*)d_in[7];
  const float* ln_b   = (const float*)d_in[8];
  const float* WiAll  = (const float*)d_in[9];
  const float* cwAll  = (const float*)d_in[10];
  const float* cbAll  = (const float*)d_in[11];
  const float* WxAll  = (const float*)d_in[12];
  const float* WdtAll = (const float*)d_in[13];
  const float* bdtAll = (const float*)d_in[14];
  const float* AlogAll= (const float*)d_in[15];
  const float* DskAll = (const float*)d_in[16];
  const float* WoAll  = (const float*)d_in[17];
  const float* ng     = (const float*)d_in[18];
  const float* nb     = (const float*)d_in[19];
  const float* aW     = (const float*)d_in[20];
  const float* ab     = (const float*)d_in[21];
  float* out = (float*)d_out;   // [16*512 day_embed | 16*1020*512 episode_ctx]

  // ---- workspace carve-up --------------------------------------------------
  char* base = (char*)d_ws; size_t off = 0;
  auto alloc = [&](size_t bytes) -> void* {
    void* r = base + off; off = (off + bytes + 255) & ~(size_t)255; return r;
  };
  _Float16* w_in16  = (_Float16*)alloc((size_t)NL*2*(2*DI)*DM * 2);
  _Float16* w_x16   = (_Float16*)alloc((size_t)NL*2*XPAD*DM   * 2);
  _Float16* w_dt16  = (_Float16*)alloc((size_t)NL*2*DI*RNK    * 2);
  _Float16* w_out16 = (_Float16*)alloc((size_t)NL*2*DM*DI     * 2);
  _Float16* w_f1    = (_Float16*)alloc((size_t)DM*DM * 2);
  _Float16* w_f2    = (_Float16*)alloc((size_t)DM*DM * 2);
  _Float16* cat16   = (_Float16*)alloc((size_t)MEP*DM * 2);
  _Float16* t1_16   = (_Float16*)alloc((size_t)MEP*DM * 2);
  _Float16* h16     = (_Float16*)alloc((size_t)BL*DM * 2);
  _Float16* xc16    = (_Float16*)alloc((size_t)BL*DI * 2);
  _Float16* dbl16   = (_Float16*)alloc((size_t)BL*XPAD * 2);
  _Float16* y16     = (_Float16*)alloc((size_t)BL*DI * 2);
  float* x    = (float*)alloc((size_t)BL*DM * 4);
  float* zfus = (float*)alloc((size_t)MEP*DM * 4);
  float* xz   = (float*)alloc((size_t)BL*1024 * 4);
  float* dblf = (float*)alloc((size_t)BL*XPAD * 4);
  float* dtf  = (float*)alloc((size_t)BL*DI * 4);
  float* xcf  = (float*)alloc((size_t)BL*DI * 4);
  float* xln  = (float*)alloc((size_t)BL*DM * 4);
  float* scr  = (float*)alloc((size_t)BL * 4);

  auto cvt = [&](const float* s, _Float16* d, long long n) {
    k_convert<<<dim3((unsigned)((n + 255) / 256)), dim3(256), 0, stream>>>(s, d, n);
  };
  auto gemm = [&](int epi, const _Float16* A, const _Float16* W, const float* bias,
                  float* Cf, _Float16* Ch, int M, int N, int K,
                  int lda, int ldw, int ldc) {
    const int waves  = (M / 16) * (N / 64);
    const dim3 g((waves + 7) / 8), blk(256);
    switch (epi) {
      case 0: gemm16<0><<<g, blk, 0, stream>>>(A, W, bias, Cf, Ch, M, N, K, lda, ldw, ldc); break;
      case 1: gemm16<1><<<g, blk, 0, stream>>>(A, W, bias, Cf, Ch, M, N, K, lda, ldw, ldc); break;
      case 2: gemm16<2><<<g, blk, 0, stream>>>(A, W, bias, Cf, Ch, M, N, K, lda, ldw, ldc); break;
      case 3: gemm16<3><<<g, blk, 0, stream>>>(A, W, bias, Cf, Ch, M, N, K, lda, ldw, ldc); break;
      default: gemm16<4><<<g, blk, 0, stream>>>(A, W, bias, Cf, Ch, M, N, K, lda, ldw, ldc); break;
    }
  };

  // ---- weight conversion to f16 -------------------------------------------
  cvt(WiAll,  w_in16,  (long long)NL*2*(2*DI)*DM);
  k_pad_xproj<<<dim3((unsigned)(((long long)NL*2*XPAD*DM + 255)/256)), dim3(256), 0, stream>>>(WxAll, w_x16);
  cvt(WdtAll, w_dt16,  (long long)NL*2*DI*RNK);
  cvt(WoAll,  w_out16, (long long)NL*2*DM*DI);
  cvt(fW1, w_f1, (long long)DM*DM);
  cvt(fW2, w_f2, (long long)DM*DM);

  // ---- fusion MLP ----------------------------------------------------------
  k_concat<<<dim3((unsigned)(((long long)MEP*DM + 255)/256)), dim3(256), 0, stream>>>(wav, rhy, cat16);
  gemm(1, cat16, w_f1, fb1, nullptr, t1_16, MEP, DM, DM, DM, DM, DM);   // GELU(z@W1^T+b1)
  gemm(2, t1_16, w_f2, fb2, zfus, nullptr, MEP, DM, DM, DM, DM, DM);    // @W2^T+b2
  k_build_x<<<dim3((unsigned)(((long long)BL*DM + 255)/256)), dim3(256), 0, stream>>>(zfus, summ, x);

  // ---- 12 bidirectional Mamba layers --------------------------------------
  const unsigned convBlocks = (unsigned)(((long long)BL*DI + 255) / 256);
  for (int l = 0; l < NL; ++l) {
    k_layernorm<<<dim3(BL/8), dim3(256), 0, stream>>>(x, ln_g + l*DM, ln_b + l*DM, nullptr, h16, BL);
    for (int dir = 0; dir < 2; ++dir) {
      const size_t ld = (size_t)(l*2 + dir);
      // in_proj: xz = h @ Wi^T   [BL,1024]
      gemm(0, h16, w_in16 + ld*(2*DI)*DM, nullptr, xz, nullptr, BL, 2*DI, DM, DM, DM, 2*DI);
      // causal/anti-causal depthwise conv + SiLU
      k_conv_silu<<<dim3(convBlocks), dim3(256), 0, stream>>>(
          xz, cwAll + ld*DI*4, cbAll + ld*DI, xcf, xc16, dir);
      // x_proj: dbl = xc @ Wx^T  [BL,192] (cols 160..191 are zero padding)
      gemm(0, xc16, w_x16 + ld*XPAD*DM, nullptr, dblf, dbl16, BL, XPAD, DM, DM, DM, XPAD);
      // dt = softplus(dbl[:, :32] @ Wdt^T + bdt)   [BL,512]
      gemm(3, dbl16, w_dt16 + ld*DI*RNK, bdtAll + ld*DI, dtf, nullptr, BL, DI, RNK, XPAD, RNK, DI);
      // selective scan (+D skip, * silu(z)) -> y16
      k_scan<<<dim3(NB, DI/128), dim3(128), 0, stream>>>(
          dblf, dtf, xcf, xz, AlogAll + ld*DI*DSTATE, DskAll + ld*DI, y16, dir);
      // out_proj accumulated into residual: x += y @ Wo^T
      gemm(4, y16, w_out16 + ld*DM*DI, nullptr, x, nullptr, BL, DM, DI, DI, DI, DM);
    }
  }

  // ---- final norm, attention pooling, outputs ------------------------------
  k_layernorm<<<dim3(BL/8), dim3(256), 0, stream>>>(x, ng, nb, xln, nullptr, BL);
  k_scores<<<dim3(BL/8), dim3(256), 0, stream>>>(xln, aW, ab, scr);
  k_pool<<<dim3(NB), dim3(256), 0, stream>>>(xln, scr, out);
  k_ctx<<<dim3((unsigned)(((long long)MEP*DM + 255)/256)), dim3(256), 0, stream>>>(xln, out + (size_t)NB*DM);
}
